// MultiHeadAttention_1262720385502
// MI455X (gfx1250) — matrix operand order repaired and checked by reading, not compiled, not run on hardware
//
#include <hip/hip_runtime.h>

// Problem constants (match reference)
#define BB   4
#define SS   1024
#define HHD  1024
#define NH   16
#define DKK  64

typedef __attribute__((ext_vector_type(16))) _Float16 v16h;
typedef __attribute__((ext_vector_type(8)))  _Float16 v8h;
typedef __attribute__((ext_vector_type(4)))  _Float16 v4h;
typedef __attribute__((ext_vector_type(8)))  float    v8f;
typedef __attribute__((ext_vector_type(4)))  unsigned u32x4;
typedef __attribute__((ext_vector_type(8)))  int      i32x8;
typedef __attribute__((ext_vector_type(4)))  int      i32x4;

static __device__ __forceinline__ v16h cat16(v8h lo, v8h hi) {
  return __builtin_shufflevector(lo, hi, 0,1,2,3,4,5,6,7,8,9,10,11,12,13,14,15);
}

static __device__ __forceinline__ v8f wmma_f16(v16h a, v16h b, v8f c) {
  // v_wmma_f32_16x16x32_f16
  return __builtin_amdgcn_wmma_f32_16x16x32_f16(false, a, false, b, (short)0, c, false, false);
}

// ---------------------------------------------------------------------------
// DPP16 butterfly reductions confined to each 16-lane half (pure VALU,
// replaces ds_bpermute-based __shfl_xor).
// ---------------------------------------------------------------------------
#if __has_builtin(__builtin_amdgcn_update_dpp)
template<int CTRL>
static __device__ __forceinline__ float dppf(float x) {
  return __int_as_float(__builtin_amdgcn_update_dpp(
      __float_as_int(x), __float_as_int(x), CTRL, 0xF, 0xF, true));
}
static __device__ __forceinline__ float rowmax16(float x) {
  x = fmaxf(x, dppf<0xB1>(x));    // quad_perm [1,0,3,2]  (xor 1)
  x = fmaxf(x, dppf<0x4E>(x));    // quad_perm [2,3,0,1]  (xor 2)
  x = fmaxf(x, dppf<0x141>(x));   // row_half_mirror      (pairs quads)
  x = fmaxf(x, dppf<0x140>(x));   // row_mirror           (pairs octets)
  return x;
}
static __device__ __forceinline__ float rowsum16(float x) {
  x += dppf<0xB1>(x);
  x += dppf<0x4E>(x);
  x += dppf<0x141>(x);
  x += dppf<0x140>(x);
  return x;
}
#else
static __device__ __forceinline__ float rowmax16(float x) {
  for (int off = 8; off >= 1; off >>= 1) x = fmaxf(x, __shfl_xor(x, off, 32));
  return x;
}
static __device__ __forceinline__ float rowsum16(float x) {
  for (int off = 8; off >= 1; off >>= 1) x += __shfl_xor(x, off, 32);
  return x;
}
#endif

// ---------------------------------------------------------------------------
// Tensor Data Mover: 2D f16 tile load (global -> LDS).
// This toolchain exposes the 6-arg builtin:
//   (uint32x4 g0, int32x8 g1, int32x4, int32x4, int32x8, i32 cpol)
// D# packed per CDNA5 ISA ch.8 (group0: count/lds/global/type=2;
// group1: data_size=2B, dims, tile dims, dim0 stride). Groups 2/3 unused (2D).
// ---------------------------------------------------------------------------
#if __has_builtin(__builtin_amdgcn_tensor_load_to_lds) && __has_builtin(__builtin_amdgcn_s_wait_tensorcnt)
#define HAS_TDM 1
static __device__ __forceinline__ void tdm_load_2d_f16(
    unsigned lds_addr, const void* gptr,
    unsigned tile_w, unsigned tile_h,
    unsigned tensor_w, unsigned tensor_h, unsigned row_stride) {
  unsigned long long ga = (unsigned long long)(size_t)gptr;
  u32x4 g0;
  g0[0] = 1u;                                         // count=1 (user D#)
  g0[1] = lds_addr;                                   // LDS byte address
  g0[2] = (unsigned)(ga & 0xffffffffu);               // global_addr[31:0]
  g0[3] = (unsigned)((ga >> 32) & 0x1ffffffu) | (2u << 30);  // addr[56:32] | type=2
  i32x8 g1;
  g1[0] = 1 << 16;                                    // data_size=1 (2 bytes); wg_mask=0
  g1[1] = (int)(tensor_w << 16);                      // tensor_dim0[15:0] @ [63:48]
  g1[2] = (int)((tensor_w >> 16) & 0xffffu) | (int)(tensor_h << 16);   // dim0 hi | dim1 lo
  g1[3] = (int)((tensor_h >> 16) & 0xffffu) | (int)(tile_w << 16);     // dim1 hi | tile_dim0
  g1[4] = (int)tile_h;                                // tile_dim1; tile_dim2=0
  g1[5] = (int)row_stride;                            // tensor_dim0_stride[31:0]
  g1[6] = 0; g1[7] = 0;
  i32x4 z4 = {0, 0, 0, 0};
  i32x8 z8 = {0, 0, 0, 0, 0, 0, 0, 0};
  __builtin_amdgcn_tensor_load_to_lds(g0, g1, z4, z4, z8, 0);
}
#else
#define HAS_TDM 0
#endif

// ---------------------------------------------------------------------------
// f32 -> f16 elementwise convert
// ---------------------------------------------------------------------------
__global__ void cvt_f32_to_f16(const float* __restrict__ in,
                               _Float16* __restrict__ out, int n) {
  int i = blockIdx.x * 256 + threadIdx.x;
  if (i < n) out[i] = (_Float16)in[i];
}

// ---------------------------------------------------------------------------
// Pack bool mask bytes -> bit mask (1 bit per key). 32 bytes -> 1 u32.
// ---------------------------------------------------------------------------
__global__ void pack_mask(const unsigned char* __restrict__ m,
                          unsigned* __restrict__ w, int nwords) {
  int i = blockIdx.x * 256 + threadIdx.x;
  if (i >= nwords) return;
  const unsigned* p = (const unsigned*)(m + (size_t)i * 32);
  unsigned bits = 0;
  #pragma unroll
  for (int j = 0; j < 8; ++j) {
    unsigned v = p[j];
    if (v & 0x000000ffu) bits |= 1u << (4 * j + 0);
    if (v & 0x0000ff00u) bits |= 1u << (4 * j + 1);
    if (v & 0x00ff0000u) bits |= 1u << (4 * j + 2);
    if (v & 0xff000000u) bits |= 1u << (4 * j + 3);
  }
  w[i] = bits;
}

// ---------------------------------------------------------------------------
// Y[M,N] = A[M,K] @ W[K,N] + bias[N]   (A,W f16; accumulate f32)
// Block: 256 threads = 8 wave32s; block tile 64(M) x 128(N), k-step 32.
// Wave (wm, wn) computes rows m0+16*wm, cols n0+64*wn .. +63 (4 WMMA tiles).
// ---------------------------------------------------------------------------
template<bool F16OUT>
__global__ __launch_bounds__(256)
void gemm_bias_wmma(const _Float16* __restrict__ A,
                    const _Float16* __restrict__ W,
                    const float* __restrict__ bias,
                    _Float16* __restrict__ outH,
                    float* __restrict__ outF,
                    int M, int N, int K) {
  __shared__ __align__(16) _Float16 As[64][32];    // [m][k]
  __shared__ __align__(16) _Float16 Bs[128][32];   // [n][k] (transposed W tile)
  const int tid  = threadIdx.x;
  const int wave = tid >> 5, lane = tid & 31;
  const int half = lane >> 4, lr = lane & 15;
  const int wm = wave & 3, wn = wave >> 2;
  const int m0 = blockIdx.y * 64;
  const int n0 = blockIdx.x * 128;

  v8f c[4] = {};
  for (int k0 = 0; k0 < K; k0 += 32) {
    __syncthreads();
    { // stage A tile: 64x32 f16, 16B per thread
      int r = tid >> 2, c8 = (tid & 3) * 8;
      *(v8h*)&As[r][c8] = *(const v8h*)&A[(size_t)(m0 + r) * K + k0 + c8];
    }
    #pragma unroll
    for (int it = 0; it < 2; ++it) { // stage W tile transposed: 32(k) x 128(n)
      int task = tid + it * 256;
      int kk = task >> 4, cg = (task & 15) * 8;
      v8h w = *(const v8h*)&W[(size_t)(k0 + kk) * N + n0 + cg];
      #pragma unroll
      for (int j = 0; j < 8; ++j) Bs[cg + j][kk] = w[j];
    }
    __syncthreads();
    v16h a = cat16(*(const v8h*)&As[wm * 16 + lr][16 * half],
                   *(const v8h*)&As[wm * 16 + lr][16 * half + 8]);
    #pragma unroll
    for (int ns = 0; ns < 4; ++ns) {
      const _Float16* bp = &Bs[wn * 64 + ns * 16 + lr][16 * half];
      v16h b = cat16(*(const v8h*)bp, *(const v8h*)(bp + 8));
      c[ns] = wmma_f16(a, b, c[ns]);
    }
  }

  #pragma unroll
  for (int ns = 0; ns < 4; ++ns) {
    const int col = n0 + wn * 64 + ns * 16 + lr;
    const float bv = bias[col];
    #pragma unroll
    for (int r = 0; r < 8; ++r) {
      int row = m0 + wm * 16 + r + 8 * half;
      float v = c[ns][r] + bv;
      if (F16OUT) outH[(size_t)row * N + col] = (_Float16)v;
      else        outF[(size_t)row * N + col] = v;
    }
  }
}

// ---------------------------------------------------------------------------
// Fused flash-style attention.  V == projected Q (reference quirk).
// Block: 128 threads = 4 waves; block owns 64 q-rows of one (b,h);
// each wave owns 16 q-rows. Keys processed 32 at a time.
// ---------------------------------------------------------------------------
__global__ __launch_bounds__(128)
void attention_wmma(const _Float16* __restrict__ Q,     // [B,S,H] projected q (also V)
                    const _Float16* __restrict__ Km,    // [B,S,H] projected k
                    const unsigned* __restrict__ maskw, // [B,S,S/32] packed bits
                    _Float16* __restrict__ O) {         // [B,S,H] merged-head out
  __shared__ __align__(16) _Float16 Qs[64][64];     // q rows (dk-contig)
  __shared__ __align__(16) _Float16 Ks[32][64];     // k rows (dk-contig)
  __shared__ __align__(16) _Float16 Vt[64][32];     // V transposed: [dk][key]
  __shared__ __align__(16) _Float16 Ps[4][16][32];  // per-wave P tile

  const int tid  = threadIdx.x;
  const int wave = tid >> 5, lane = tid & 31;
  const int half = lane >> 4, lr = lane & 15;
  const int q0 = blockIdx.x * 64;
  const int h  = blockIdx.y;
  const int b  = blockIdx.z;
  const size_t bh = ((size_t)b * SS) * HHD + (size_t)h * DKK;

#if HAS_TDM
  if (wave == 0) { // TDM: 64x64 q tile -> Qs
    tdm_load_2d_f16((unsigned)(size_t)&Qs[0][0], &Q[bh + (size_t)q0 * HHD],
                    64, 64, HHD, SS, HHD);
    __builtin_amdgcn_s_wait_tensorcnt(0);
  }
#else
  { // stage 64x64 q tile manually
    int r = tid >> 1, cs = (tid & 1) * 32;
    const _Float16* g = &Q[bh + (size_t)(q0 + r) * HHD + cs];
    #pragma unroll
    for (int j = 0; j < 4; ++j)
      *(v8h*)&Qs[r][cs + 8 * j] = *(const v8h*)&g[8 * j];
  }
#endif
  __syncthreads();

  const int qr0 = wave * 16;
  // persistent A fragments of q (dk 0..31 and 32..63)
  v16h aq_lo = cat16(*(const v8h*)&Qs[qr0 + lr][16 * half],
                     *(const v8h*)&Qs[qr0 + lr][16 * half + 8]);
  v16h aq_hi = cat16(*(const v8h*)&Qs[qr0 + lr][32 + 16 * half],
                     *(const v8h*)&Qs[qr0 + lr][40 + 16 * half]);

  float m[8], l[8];
  #pragma unroll
  for (int r = 0; r < 8; ++r) { m[r] = -__builtin_inff(); l[r] = 0.f; }
  v8f acc[4] = {};   // 16 rows x 64 dk output accumulator (4 N-subtiles)

  for (int kt = 0; kt < SS; kt += 32) {
    __syncthreads();
#if HAS_TDM
    if (wave == 0) { // TDM: 32x64 k tile -> Ks
      tdm_load_2d_f16((unsigned)(size_t)&Ks[0][0], &Km[bh + (size_t)kt * HHD],
                      64, 32, HHD, SS, HHD);
    }
#else
    { // stage K tile rows kt..kt+31 manually
      int kk = tid >> 2, cs = (tid & 3) * 16;
      const _Float16* g = &Km[bh + (size_t)(kt + kk) * HHD + cs];
      *(v8h*)&Ks[kk][cs]     = *(const v8h*)&g[0];
      *(v8h*)&Ks[kk][cs + 8] = *(const v8h*)&g[8];
    }
#endif
    { // stage V tile transposed (V = Q rows kt..kt+31)
      int kk = tid & 31, cg = (tid >> 5) * 16;
      const _Float16* g = &Q[bh + (size_t)(kt + kk) * HHD + cg];
      v8h a0 = *(const v8h*)&g[0];
      v8h a1 = *(const v8h*)&g[8];
      #pragma unroll
      for (int j = 0; j < 8; ++j) { Vt[cg + j][kk] = a0[j]; Vt[cg + 8 + j][kk] = a1[j]; }
    }
#if HAS_TDM
    if (wave == 0) __builtin_amdgcn_s_wait_tensorcnt(0);
#endif
    __syncthreads();

    // scores: 16 q-rows x 32 keys = two 16x16 C tiles, dk reduced in 2 chunks
    v8f s[2];
    #pragma unroll
    for (int t = 0; t < 2; ++t) {
      const _Float16* kp = &Ks[16 * t + lr][0];   // B col = key, K = dk
      v16h b_lo = cat16(*(const v8h*)&kp[16 * half],      *(const v8h*)&kp[16 * half + 8]);
      v16h b_hi = cat16(*(const v8h*)&kp[32 + 16 * half], *(const v8h*)&kp[32 + 16 * half + 8]);
      v8f z = {};
      z = wmma_f16(aq_lo, b_lo, z);
      z = wmma_f16(aq_hi, b_hi, z);
      s[t] = z;
    }

    // scale (1/sqrt(64)) + mask bits + online softmax update
    const size_t mwbase = ((size_t)b * SS) * (SS / 32) + (kt >> 5);
    #pragma unroll
    for (int r = 0; r < 8; ++r) {
      int row = q0 + qr0 + r + 8 * half;
      unsigned mw = maskw[mwbase + (size_t)row * (SS / 32)];
      float s0 = s[0][r] * 0.125f;
      float s1 = s[1][r] * 0.125f;
      if ((mw >> lr) & 1u)        s0 = -1e9f;
      if ((mw >> (16 + lr)) & 1u) s1 = -1e9f;
      float mt = rowmax16(fmaxf(s0, s1));
      float mnew = fmaxf(m[r], mt);
      float corr = __expf(m[r] - mnew);
      float p0 = __expf(s0 - mnew);
      float p1 = __expf(s1 - mnew);
      float rs = rowsum16(p0 + p1);
      l[r] = l[r] * corr + rs;
      m[r] = mnew;
      #pragma unroll
      for (int ns = 0; ns < 4; ++ns) acc[ns][r] *= corr;
      // C layout -> LDS (row-major P tile) for A-fragment reload
      Ps[wave][r + 8 * half][lr]      = (_Float16)p0;
      Ps[wave][r + 8 * half][16 + lr] = (_Float16)p1;
    }
    __syncthreads();

    // acc += P(16x32) @ V(32x64)
    v16h ap = cat16(*(const v8h*)&Ps[wave][lr][16 * half],
                    *(const v8h*)&Ps[wave][lr][16 * half + 8]);
    #pragma unroll
    for (int ns = 0; ns < 4; ++ns) {
      const _Float16* vp = &Vt[ns * 16 + lr][16 * half];  // B col = dk, K = key
      v16h bv = cat16(*(const v8h*)&vp[0], *(const v8h*)&vp[8]);
      acc[ns] = wmma_f16(ap, bv, acc[ns]);
    }
  }

  // epilogue: normalize, apply row_keep = ~mask[b,row,0], write merged heads
  #pragma unroll
  for (int r = 0; r < 8; ++r) {
    int row = q0 + qr0 + r + 8 * half;
    float keep = (maskw[((size_t)b * SS + row) * (SS / 32)] & 1u) ? 0.f : 1.f;
    float invl = keep / l[r];
    #pragma unroll
    for (int ns = 0; ns < 4; ++ns)
      O[bh + (size_t)row * HHD + ns * 16 + lr] = (_Float16)(acc[ns][r] * invl);
  }
}

// ---------------------------------------------------------------------------
extern "C" void kernel_launch(void* const* d_in, const int* in_sizes, int n_in,
                              void* d_out, int out_size, void* d_ws, size_t ws_size,
                              hipStream_t stream) {
  (void)in_sizes; (void)n_in; (void)out_size; (void)ws_size;
  const float* query  = (const float*)d_in[0];
  const float* key_in = (const float*)d_in[1];
  const unsigned char* mask = (const unsigned char*)d_in[2];  // numpy bool
  const float* Wq = (const float*)d_in[3];
  const float* bq = (const float*)d_in[4];
  const float* Wk = (const float*)d_in[5];
  const float* bk = (const float*)d_in[6];
  const float* Wp = (const float*)d_in[7];
  const float* bp = (const float*)d_in[8];
  float* out = (float*)d_out;

  const int MSH = BB * SS * HHD;   // 4,194,304
  const int HH2 = HHD * HHD;       // 1,048,576
  const int NW  = BB * SS * (SS / 32);  // 131,072 packed mask words

  _Float16* ws   = (_Float16*)d_ws;
  _Float16* Xq16 = ws;                 // query f16
  _Float16* Xk16 = Xq16 + MSH;         // key_in f16
  _Float16* q16  = Xk16 + MSH;         // projected q (also V)
  _Float16* k16  = q16  + MSH;         // projected k
  _Float16* a16  = k16  + MSH;         // attention output, merged heads
  _Float16* Wq16 = a16  + MSH;
  _Float16* Wk16 = Wq16 + HH2;
  _Float16* Wp16 = Wk16 + HH2;
  unsigned* maskw = (unsigned*)(Wp16 + HH2);

  cvt_f32_to_f16<<<(MSH + 255) / 256, 256, 0, stream>>>(query,  Xq16, MSH);
  cvt_f32_to_f16<<<(MSH + 255) / 256, 256, 0, stream>>>(key_in, Xk16, MSH);
  cvt_f32_to_f16<<<(HH2 + 255) / 256, 256, 0, stream>>>(Wq, Wq16, HH2);
  cvt_f32_to_f16<<<(HH2 + 255) / 256, 256, 0, stream>>>(Wk, Wk16, HH2);
  cvt_f32_to_f16<<<(HH2 + 255) / 256, 256, 0, stream>>>(Wp, Wp16, HH2);
  pack_mask<<<(NW + 255) / 256, 256, 0, stream>>>(mask, maskw, NW);

  dim3 gg(HHD / 128, (BB * SS) / 64);  // (8, 64)
  gemm_bias_wmma<true ><<<gg, 256, 0, stream>>>(Xq16, Wq16, bq, q16, nullptr, BB * SS, HHD, HHD);
  gemm_bias_wmma<true ><<<gg, 256, 0, stream>>>(Xk16, Wk16, bk, k16, nullptr, BB * SS, HHD, HHD);

  dim3 ga(SS / 64, NH, BB);            // (16, 16, 4)
  attention_wmma<<<ga, 128, 0, stream>>>(q16, k16, maskw, a16);

  gemm_bias_wmma<false><<<gg, 256, 0, stream>>>(a16, Wp16, bp, nullptr, out, BB * SS, HHD, HHD);
}
